// CambaBlock_38603166057070
// MI455X (gfx1250) — matrix operand order repaired and checked by reading, not compiled, not run on hardware
//
#include <hip/hip_runtime.h>
#include <math.h>

typedef __attribute__((ext_vector_type(16))) _Float16 v16h;
typedef __attribute__((ext_vector_type(8)))  _Float16 v8h;
typedef __attribute__((ext_vector_type(8)))  float    v8f;
typedef __attribute__((ext_vector_type(4)))  float    v4f;

#define TOK    32768   // b*h*w tokens
#define BATCH  8
#define CHN    64
#define HGT    64
#define WID    64
#define LSEQ   4096    // h*w per batch
#define DI     128
#define DSTATE 16
#define DTR    4

// ---------------------------------------------------------------------------
// Weight packer: f32 W[K x N] row-major  ->  f16 WP[Npad x Kpad] column-major
// (K contiguous per column), zero padded. Lets the GEMM B-fragment be two
// unguarded b128 loads per lane with no bounds checks.
// ---------------------------------------------------------------------------
__global__ void k_pack_w(const float* __restrict__ W, _Float16* __restrict__ WP,
                         int K, int N, int Kpad, int Npad)
{
    int idx = blockIdx.x * blockDim.x + threadIdx.x;
    if (idx >= Kpad * Npad) return;
    int k = idx % Kpad, n = idx / Kpad;
    WP[(size_t)n * Kpad + k] =
        (_Float16)((k < K && n < N) ? W[(size_t)k * N + n] : 0.0f);
}

// ---------------------------------------------------------------------------
// WMMA GEMM:  out[M x N] = act(A[M x K] @ W[K x N] + bias) + res
// A: f32, row stride lda (multiple of 4).  WP: packed f16 (see k_pack_w).
// One wave per 16x16 C tile; K-loop in steps of 32.
// A fragment  = 4 unguarded float4 loads (two 8-float runs) + cvt to f16.
// B fragment  = 2 unguarded b128 loads of 8 halves each (contiguous).
// KGUARD=true only for the dt GEMM (K=4): branchless clamp + select.
// ---------------------------------------------------------------------------
template<int ACT, bool KGUARD>  // ACT: 0=none 1=gelu 2=leaky0.2 3=softplus
__global__ void k_gemm_wmma(const float* __restrict__ A, const _Float16* __restrict__ WP,
                            const float* __restrict__ bias, const float* __restrict__ res,
                            float* __restrict__ out,
                            int M, int K, int Kpad, int N, int lda, int ldo)
{
    const int lane = threadIdx.x & 31;
    const int wave = blockIdx.x * (blockDim.x >> 5) + (threadIdx.x >> 5);
    const int ntn  = (N + 15) >> 4;
    if (wave >= (M >> 4) * ntn) return;
    const int tm   = wave / ntn, tn = wave % ntn;
    const int row0 = tm << 4,  col0 = tn << 4;
    const int hf   = lane >> 4;          // 16-lane half of the wave32
    const int arow = row0 + (lane & 15);
    const int bcol = col0 + (lane & 15);

    const float*    Arow = A  + (size_t)arow * lda;
    const _Float16* Bcol = WP + (size_t)bcol * Kpad;   // bcol < Npad by packing

    v8f acc = {};
    for (int k0 = 0; k0 < Kpad; k0 += 32) {
        v16h av, bv;
        if (!KGUARD) {
            // Two contiguous 8-float runs: k0+hf*8.. and k0+16+hf*8..
            const v4f* pa0 = (const v4f*)(Arow + k0 + hf * 8);
            const v4f* pa1 = (const v4f*)(Arow + k0 + 16 + hf * 8);
            v4f a0 = pa0[0], a1 = pa0[1], a2 = pa1[0], a3 = pa1[1];
#pragma unroll
            for (int i = 0; i < 4; ++i) {
                av[i]      = (_Float16)a0[i];
                av[4 + i]  = (_Float16)a1[i];
                av[8 + i]  = (_Float16)a2[i];
                av[12 + i] = (_Float16)a3[i];
            }
            if (k0 + 32 < Kpad)
                __builtin_prefetch(Arow + k0 + 32 + hf * 8, 0, 1);
        } else {
            // Branchless: clamp address, select value (no exec-mask branches)
#pragma unroll
            for (int i = 0; i < 8; ++i) {
                const int ka = k0 + hf * 8 + i;
                const int kb = k0 + 16 + hf * 8 + i;
                float va = Arow[(ka < K) ? ka : 0];
                float vb = Arow[(kb < K) ? kb : 0];
                av[i]     = (_Float16)((ka < K) ? va : 0.0f);
                av[8 + i] = (_Float16)((kb < K) ? vb : 0.0f);
            }
        }
        // B: two 8-half runs at k0 + hf*8 and k0 + 16 + hf*8
        const v8h* pb0 = (const v8h*)(Bcol + k0 + hf * 8);
        const v8h* pb1 = (const v8h*)(Bcol + k0 + 16 + hf * 8);
        v8h b0 = pb0[0], b1 = pb1[0];
#pragma unroll
        for (int i = 0; i < 8; ++i) { bv[i] = b0[i]; bv[8 + i] = b1[i]; }

        acc = __builtin_amdgcn_wmma_f32_16x16x32_f16(false, av, false, bv,
                                                     (short)0, acc, false, false);
    }

    if (bcol < N) {
        const int rbase = row0 + hf * 8;  // C/D: vgpr v -> row hf*8+v
#pragma unroll
        for (int v = 0; v < 8; ++v) {
            float val = acc[v];
            if (bias) val += bias[bcol];
            if (ACT == 1)      val = 0.5f * val * (1.0f + erff(val * 0.70710678f));
            else if (ACT == 2) val = (val > 0.0f) ? val : 0.2f * val;
            else if (ACT == 3) val = (val > 20.0f) ? val : log1pf(expf(val));
            const size_t o = (size_t)(rbase + v) * ldo + bcol;
            if (res) val += res[o];
            out[o] = val;
        }
    }
}

// ---------------------------------------------------------------------------
// NCHW -> NHWC transpose + LayerNorm (keeps raw NHWC copy for residual)
// ---------------------------------------------------------------------------
__global__ void k_trans_ln(const float* __restrict__ x, const float* __restrict__ g,
                           const float* __restrict__ bt, float* __restrict__ xh,
                           float* __restrict__ xln)
{
    int t = blockIdx.x * blockDim.x + threadIdx.x;
    if (t >= TOK) return;
    int b = t / LSEQ, hw = t % LSEQ;
    float v[CHN], m = 0.0f;
#pragma unroll
    for (int c = 0; c < CHN; ++c) { v[c] = x[((size_t)(b * CHN + c)) * LSEQ + hw]; m += v[c]; }
    m *= (1.0f / CHN);
    float var = 0.0f;
#pragma unroll
    for (int c = 0; c < CHN; ++c) { float d = v[c] - m; var += d * d; }
    var *= (1.0f / CHN);
    float inv = rsqrtf(var + 1e-5f);
#pragma unroll
    for (int c = 0; c < CHN; ++c) {
        xh [(size_t)t * CHN + c] = v[c];
        xln[(size_t)t * CHN + c] = (v[c] - m) * inv * g[c] + bt[c];
    }
}

__global__ void k_ln(const float* __restrict__ in, const float* __restrict__ g,
                     const float* __restrict__ bt, float* __restrict__ out)
{
    int t = blockIdx.x * blockDim.x + threadIdx.x;
    if (t >= TOK) return;
    float v[CHN], m = 0.0f;
#pragma unroll
    for (int c = 0; c < CHN; ++c) { v[c] = in[(size_t)t * CHN + c]; m += v[c]; }
    m *= (1.0f / CHN);
    float var = 0.0f;
#pragma unroll
    for (int c = 0; c < CHN; ++c) { float d = v[c] - m; var += d * d; }
    var *= (1.0f / CHN);
    float inv = rsqrtf(var + 1e-5f);
#pragma unroll
    for (int c = 0; c < CHN; ++c)
        out[(size_t)t * CHN + c] = (v[c] - m) * inv * g[c] + bt[c];
}

// ---------------------------------------------------------------------------
// Depthwise 3x3 conv, SAME padding, NHWC, optional activation + residual
// ---------------------------------------------------------------------------
template<int ACT>
__global__ void k_dwconv3x3(const float* __restrict__ in, const float* __restrict__ w,
                            const float* __restrict__ res, float* __restrict__ out, int C)
{
    size_t idx = (size_t)blockIdx.x * blockDim.x + threadIdx.x;
    if (idx >= (size_t)TOK * C) return;
    int c = (int)(idx % C);
    int t = (int)(idx / C);
    int b = t / LSEQ, hw = t % LSEQ, y = hw / WID, x = hw % WID;
    float s = 0.0f;
#pragma unroll
    for (int ky = 0; ky < 3; ++ky) {
        int yy = y + ky - 1; if (yy < 0 || yy >= HGT) continue;
#pragma unroll
        for (int kx = 0; kx < 3; ++kx) {
            int xx = x + kx - 1; if (xx < 0 || xx >= WID) continue;
            s += in[((size_t)b * LSEQ + yy * WID + xx) * C + c] * w[(ky * 3 + kx) * C + c];
        }
    }
    if (ACT == 1)      s = 0.5f * s * (1.0f + erff(s * 0.70710678f));
    else if (ACT == 2) s = (s > 0.0f) ? s : 0.2f * s;
    if (res) s += res[idx];
    out[idx] = s;
}

// ---------------------------------------------------------------------------
// Causal depthwise conv1d (D_CONV=4) over flattened sequence, + bias + SiLU.
// Source is xz[..., :DI] with row stride 256.
// ---------------------------------------------------------------------------
__global__ void k_conv1d_silu(const float* __restrict__ xz, const float* __restrict__ w,
                              const float* __restrict__ bias, float* __restrict__ xi)
{
    size_t idx = (size_t)blockIdx.x * blockDim.x + threadIdx.x;
    if (idx >= (size_t)BATCH * LSEQ * DI) return;
    int c = (int)(idx % DI);
    int t = (int)(idx / DI);
    int b = t / LSEQ, tt = t % LSEQ;
    float s = bias[c];
#pragma unroll
    for (int j = 0; j < 4; ++j) {
        int ts = tt - 3 + j;
        if (ts >= 0) s += xz[((size_t)b * LSEQ + ts) * 256 + c] * w[j * DI + c];
    }
    s = s / (1.0f + expf(-s));     // SiLU
    xi[idx] = s;
}

// ---------------------------------------------------------------------------
// Selective scan. One wave = two (b,d) channels (16 lanes each), lane = state.
// h = exp(dt*A)*h + dt*xi*B ; y = sum_s h*C + D*xi ; y *= silu(z).
// __shfl_xor masks 1,2,4,8 reduce within each 16-lane half of the wave32.
// ---------------------------------------------------------------------------
__global__ void k_scan(const float* __restrict__ dt, const float* __restrict__ xi,
                       const float* __restrict__ dbc, const float* __restrict__ xz,
                       const float* __restrict__ Alog, const float* __restrict__ Dp,
                       float* __restrict__ ys)
{
    const int lane = threadIdx.x & 31;
    const int wave = blockIdx.x * (blockDim.x >> 5) + (threadIdx.x >> 5);
    const int hf   = lane >> 4;
    const int s    = lane & 15;
    const int ch   = wave * 2 + hf;
    if (ch >= BATCH * DI) return;
    const int b = ch / DI, d = ch % DI;
    const float As = -expf(Alog[d * DSTATE + s]);
    const float Dd = Dp[d];
    float h = 0.0f;
    for (int t = 0; t < LSEQ; ++t) {
        const size_t tok = (size_t)b * LSEQ + t;
        const float dtv = dt[tok * DI + d];
        const float xv  = xi[tok * DI + d];
        const float Bs  = dbc[tok * 36 + DTR + s];
        const float Cs  = dbc[tok * 36 + DTR + DSTATE + s];
        h = expf(dtv * As) * h + dtv * xv * Bs;
        float p = h * Cs;
#pragma unroll
        for (int off = 1; off < 16; off <<= 1) p += __shfl_xor(p, off, 32);
        if (s == 0) {
            float yv = p + Dd * xv;
            const float z = xz[tok * 256 + DI + d];
            yv *= z / (1.0f + expf(-z));          // y * silu(z)
            ys[tok * DI + d] = yv;
        }
    }
}

__global__ void k_add_inplace(float* __restrict__ a, const float* __restrict__ b, size_t n)
{
    size_t i = (size_t)blockIdx.x * blockDim.x + threadIdx.x;
    if (i < n) a[i] += b[i];
}

__global__ void k_trans_out(const float* __restrict__ in, float* __restrict__ out)
{
    int t = blockIdx.x * blockDim.x + threadIdx.x;
    if (t >= TOK) return;
    int b = t / LSEQ, hw = t % LSEQ;
#pragma unroll
    for (int c = 0; c < CHN; ++c)
        out[((size_t)(b * CHN + c)) * LSEQ + hw] = in[(size_t)t * CHN + c];
}

// ---------------------------------------------------------------------------
extern "C" void kernel_launch(void* const* d_in, const int* in_sizes, int n_in,
                              void* d_out, int out_size, void* d_ws, size_t ws_size,
                              hipStream_t stream)
{
    (void)in_sizes; (void)n_in; (void)out_size; (void)ws_size;
    const float* x        = (const float*)d_in[0];
    const float* ln1_g    = (const float*)d_in[1];
    const float* ln1_b    = (const float*)d_in[2];
    const float* vin_w1   = (const float*)d_in[3];
    const float* vin_dw   = (const float*)d_in[4];
    const float* vin_w2   = (const float*)d_in[5];
    const float* vout_dw1 = (const float*)d_in[6];
    const float* vout_dw2 = (const float*)d_in[7];
    const float* ssm_in_w = (const float*)d_in[8];
    const float* conv_w   = (const float*)d_in[9];
    const float* conv_b   = (const float*)d_in[10];
    const float* xproj_w  = (const float*)d_in[11];
    const float* dt_w     = (const float*)d_in[12];
    const float* dt_b     = (const float*)d_in[13];
    const float* Alog     = (const float*)d_in[14];
    const float* Dp       = (const float*)d_in[15];
    const float* out_w    = (const float*)d_in[16];
    const float* ln2_g    = (const float*)d_in[17];
    const float* ln2_b    = (const float*)d_in[18];
    const float* ff_w1    = (const float*)d_in[19];
    const float* ff_dw    = (const float*)d_in[20];
    const float* ff_w2    = (const float*)d_in[21];
    float* outp = (float*)d_out;

    float* ws = (float*)d_ws;
    size_t off = 0;
    auto alloc = [&](size_t nfloats) { float* p = ws + off; off += nfloats; return p; };
    auto allocH = [&](size_t nhalf) { return (_Float16*)alloc((nhalf + 1) / 2); };

    float* XH  = alloc((size_t)TOK * CHN);
    float* B1  = alloc((size_t)TOK * CHN);
    float* C1  = alloc((size_t)TOK * CHN);
    float* D1  = alloc((size_t)TOK * CHN);
    float* XZ  = alloc((size_t)TOK * 256);
    float* XI  = alloc((size_t)TOK * DI);
    float* DBC = alloc((size_t)TOK * 36);
    float* DT  = alloc((size_t)TOK * DI);
    float* YS  = alloc((size_t)TOK * DI);
    float* HID = alloc((size_t)TOK * 256);

    // packed f16 weights (Npad x Kpad, column-major, zero-padded)
    _Float16* W1p  = allocH(64  * 64);    // vin_w1   K=64  N=64
    _Float16* W2p  = allocH(64  * 64);    // vin_w2
    _Float16* WINp = allocH(256 * 64);    // ssm_in_w K=64  N=256
    _Float16* WXPp = allocH(48  * 128);   // xproj_w  K=128 N=36 -> Npad=48
    _Float16* WDTp = allocH(128 * 32);    // dt_w     K=4   N=128 -> Kpad=32
    _Float16* WOUp = allocH(64  * 128);   // out_w    K=128 N=64
    _Float16* WF1p = allocH(256 * 64);    // ff_w1    K=64  N=256
    _Float16* WF2p = allocH(64  * 256);   // ff_w2    K=256 N=64

    const int TB = 256;
    const int tokBlocks = (TOK + TB - 1) / TB;

    auto pack = [&](const float* W, _Float16* WP, int K, int N) {
        int Kpad = (K + 31) & ~31, Npad = (N + 15) & ~15;
        int n = Kpad * Npad;
        k_pack_w<<<(n + TB - 1) / TB, TB, 0, stream>>>(W, WP, K, N, Kpad, Npad);
    };
    pack(vin_w1,   W1p,  64, 64);
    pack(vin_w2,   W2p,  64, 64);
    pack(ssm_in_w, WINp, 64, 256);
    pack(xproj_w,  WXPp, 128, 36);
    pack(dt_w,     WDTp, 4, 128);
    pack(out_w,    WOUp, 128, 64);
    pack(ff_w1,    WF1p, 64, 256);
    pack(ff_w2,    WF2p, 256, 64);

    auto gemm = [&](int act, bool kguard, const float* A, const _Float16* WP,
                    const float* bias, const float* res, float* o,
                    int K, int N, int lda, int ldo) {
        int Kpad   = (K + 31) & ~31;
        int tiles  = (TOK / 16) * ((N + 15) / 16);
        int blocks = (tiles + 3) / 4;         // 4 waves (128 threads) / block
        if (!kguard) {
            switch (act) {
            case 0: k_gemm_wmma<0,false><<<blocks,128,0,stream>>>(A,WP,bias,res,o,TOK,K,Kpad,N,lda,ldo); break;
            case 2: k_gemm_wmma<2,false><<<blocks,128,0,stream>>>(A,WP,bias,res,o,TOK,K,Kpad,N,lda,ldo); break;
            default:k_gemm_wmma<1,false><<<blocks,128,0,stream>>>(A,WP,bias,res,o,TOK,K,Kpad,N,lda,ldo); break;
            }
        } else {
            k_gemm_wmma<3,true><<<blocks,128,0,stream>>>(A,WP,bias,res,o,TOK,K,Kpad,N,lda,ldo);
        }
    };

    // 1) transpose + LN1:  XH = x (NHWC), B1 = LN(x)
    k_trans_ln<<<tokBlocks, TB, 0, stream>>>(x, ln1_g, ln1_b, XH, B1);
    // 2) conv1x1 vin_w1
    gemm(0, false, B1, W1p, nullptr, nullptr, C1, 64, 64, 64, 64);
    // 3) dwconv3x3 + gelu
    k_dwconv3x3<1><<<(TOK * 64) / TB, TB, 0, stream>>>(C1, vin_dw, nullptr, D1, 64);
    // 4) conv1x1 vin_w2  -> x0 in C1
    gemm(0, false, D1, W2p, nullptr, nullptr, C1, 64, 64, 64, 64);
    // 5) xz = x0 @ ssm_in_w  (64 -> 256)
    gemm(0, false, C1, WINp, nullptr, nullptr, XZ, 64, 256, 64, 256);
    // 6) xi = silu(causal_dwconv1d(xz[:, :128]))
    k_conv1d_silu<<<(TOK * DI) / TB, TB, 0, stream>>>(XZ, conv_w, conv_b, XI);
    // 7) dbc = xi @ xproj_w  (128 -> 36)
    gemm(0, false, XI, WXPp, nullptr, nullptr, DBC, 128, 36, 128, 36);
    // 8) dt = softplus(dbc[:, :4] @ dt_w + dt_b)  (K=4 guarded path)
    gemm(3, true, DBC, WDTp, dt_b, nullptr, DT, 4, 128, 36, 128);
    // 9) selective scan + D*xi + silu(z) gating  (512 waves)
    k_scan<<<64, 256, 0, stream>>>(DT, XI, DBC, XZ, Alog, Dp, YS);
    // 10) y = ys @ out_w + x0
    gemm(0, false, YS, WOUp, nullptr, C1, B1, 128, 64, 128, 64);
    // 11-12) vout: dw2(gelu(dw1(y))) + y  -> C1
    k_dwconv3x3<1><<<(TOK * 64) / TB, TB, 0, stream>>>(B1, vout_dw1, nullptr, D1, 64);
    k_dwconv3x3<0><<<(TOK * 64) / TB, TB, 0, stream>>>(D1, vout_dw2, B1, C1, 64);
    // 13) x = y2 + x  (into XH)
    k_add_inplace<<<(TOK * 64) / TB, TB, 0, stream>>>(XH, C1, (size_t)TOK * 64);
    // 14) LN2
    k_ln<<<tokBlocks, TB, 0, stream>>>(XH, ln2_g, ln2_b, B1);
    // 15) leaky(t @ ff_w1)  (64 -> 256)
    gemm(2, false, B1, WF1p, nullptr, nullptr, XZ, 64, 256, 64, 256);
    // 16) leaky(dwconv3x3) on 256 channels
    k_dwconv3x3<2><<<(TOK * 256) / TB, TB, 0, stream>>>(XZ, ff_dw, nullptr, HID, 256);
    // 17) x = HID @ ff_w2 + x
    gemm(0, false, HID, WF2p, nullptr, XH, B1, 256, 64, 256, 64);
    // 18) NHWC -> NCHW output
    k_trans_out<<<tokBlocks, TB, 0, stream>>>(B1, outp);
}
